// DynamicFusionGraph_15522011807919
// MI455X (gfx1250) — matrix operand order repaired and checked by reading, not compiled, not run on hardware
//
#include <hip/hip_runtime.h>
#include <stdint.h>

// ---------------------------------------------------------------------------
// DynamicFusionGraph on MI455X (gfx1250): 13 GEMMs, bf16 WMMA w/ f32 accum.
// - A operand: "virtual concat" of eff-scaled segments, resolved uniformly in
//   SGPRs per K-tile (segment bounds are multiples of 256).
// - Software pipeline: global loads for tile k+1 issue while tile k's WMMAs
//   run; last iteration peeled so the prefetch is unconditional.
// - fp32->bf16 via native packed converts (__builtin_convertvector -> RNE).
// ---------------------------------------------------------------------------

typedef __bf16 bf16x2  __attribute__((ext_vector_type(2)));
typedef __bf16 bf16x8  __attribute__((ext_vector_type(8)));
typedef __bf16 bf16x16 __attribute__((ext_vector_type(16)));
typedef float  f32x2   __attribute__((ext_vector_type(2)));
typedef float  f32x8   __attribute__((ext_vector_type(8)));

struct Seg { const float* ptr; int cols; int scale_col; };
struct SegList { Seg s[7]; int n; };

#define BM 128
#define BN 128
#define BK 32
#define LDSS 40   // bf16 units per LDS row (80 B): keeps 16B alignment, spreads banks

__global__ __launch_bounds__(256)
void dfg_gemm_bf16_wmma(SegList segs, const float* __restrict__ W,
                        const float* __restrict__ bias, float* __restrict__ C,
                        const float* __restrict__ eff, int N, int K) {
  __shared__ unsigned short Alds[BM * LDSS];  // A tile, row-major [m][k] bf16
  __shared__ unsigned short Blds[BN * LDSS];  // W tile transposed [n][k] bf16

  const int tid  = threadIdx.x;
  const int lane = tid & 31;
  const int wid  = tid >> 5;
  const int wm   = wid >> 2;      // 0..1 : 64 rows per wave
  const int wn   = wid & 3;       // 0..3 : 32 cols per wave
  const int l16  = lane & 15;
  const int h    = lane >> 4;     // half-wave select (ISA frag layout)

  const int m0 = blockIdx.y * BM;
  const int n0 = blockIdx.x * BN;

  // A-fill mapping: thread owns one 16-wide K-run of one row
  const int a_row  = tid >> 1;              // 0..127
  const int a_kh   = (tid & 1) * 16;        // 0 or 16
  const int a_rowg = m0 + a_row;

  // Cumulative segment start offsets (uniform, SGPR-resident)
  int seg_start[7];
  {
    int c = 0;
    #pragma unroll
    for (int i = 0; i < 7; ++i) {
      seg_start[i] = c;
      if (i < segs.n) c += segs.s[i].cols;
    }
  }

  const bool nvec = ((N & 3) == 0) && (n0 + BN <= N);

  // Staging registers for the in-flight K-tile
  float4 va[4];
  float  ascale;
  float  wv[16];

  // Branchless, uniform segment resolve + A-tile global load (float4 x4)
  auto loadA = [&](int k0) {
    const float* p = segs.s[0].ptr;
    int cols = segs.s[0].cols, sc = segs.s[0].scale_col, base = 0;
    #pragma unroll
    for (int i = 1; i < 7; ++i) {
      if (i < segs.n && k0 >= seg_start[i]) {   // uniform -> s_cselect chain
        p = segs.s[i].ptr; cols = segs.s[i].cols;
        sc = segs.s[i].scale_col; base = seg_start[i];
      }
    }
    const float4* s4 =
        (const float4*)(p + (size_t)a_rowg * cols + (k0 + a_kh - base));
    va[0] = s4[0]; va[1] = s4[1]; va[2] = s4[2]; va[3] = s4[3];
    ascale = 1.0f;
    if (sc >= 0) ascale = eff[(size_t)a_rowg * 19 + sc];
  };

  // B-tile global load: W[k][n] fp32, coalesced along n
  auto loadB = [&](int k0) {
    #pragma unroll
    for (int i = 0; i < 4; ++i) {
      int idx = tid + i * 256;         // (32 k) x (32 groups of 4 n)
      int kl  = idx >> 5;
      int nl  = (idx & 31) * 4;
      int ng  = n0 + nl;
      const float* wp = W + (size_t)(k0 + kl) * N + ng;
      float w0 = 0.f, w1 = 0.f, w2 = 0.f, w3 = 0.f;
      if (nvec) {
        float4 w = *(const float4*)wp;
        w0 = w.x; w1 = w.y; w2 = w.z; w3 = w.w;
      } else {
        if (ng + 0 < N) w0 = wp[0];
        if (ng + 1 < N) w1 = wp[1];
        if (ng + 2 < N) w2 = wp[2];
        if (ng + 3 < N) w3 = wp[3];
      }
      wv[4 * i + 0] = w0; wv[4 * i + 1] = w1;
      wv[4 * i + 2] = w2; wv[4 * i + 3] = w3;
    }
  };

  // fp32 -> bf16 (native packed cvt, RNE) + LDS store of the staged tile
  auto storeTiles = [&]() {
    float f[16] = {va[0].x, va[0].y, va[0].z, va[0].w,
                   va[1].x, va[1].y, va[1].z, va[1].w,
                   va[2].x, va[2].y, va[2].z, va[2].w,
                   va[3].x, va[3].y, va[3].z, va[3].w};
    unsigned int* dst = (unsigned int*)&Alds[a_row * LDSS + a_kh];
    #pragma unroll
    for (int j = 0; j < 8; ++j) {
      f32x2 pr;
      pr[0] = f[2 * j] * ascale;
      pr[1] = f[2 * j + 1] * ascale;
      bf16x2 pb = __builtin_convertvector(pr, bf16x2);
      dst[j] = __builtin_bit_cast(unsigned int, pb);
    }
    #pragma unroll
    for (int i = 0; i < 4; ++i) {
      int idx = tid + i * 256;
      int kl  = idx >> 5;
      int nl  = (idx & 31) * 4;
      #pragma unroll
      for (int j = 0; j < 4; ++j)
        Blds[(nl + j) * LDSS + kl] =
            __builtin_bit_cast(unsigned short, (__bf16)wv[4 * i + j]);
    }
  };

  f32x8 acc[4][2];
  #pragma unroll
  for (int mt = 0; mt < 4; ++mt)
    #pragma unroll
    for (int nt = 0; nt < 2; ++nt)
      #pragma unroll
      for (int r = 0; r < 8; ++r) acc[mt][nt][r] = 0.0f;

  // ---- fragment loads (ISA VGPR layout) + 4x2 WMMAs from current LDS tile --
  auto compute = [&]() {
    bf16x16 afrag[4];
    #pragma unroll
    for (int mt = 0; mt < 4; ++mt) {
      int r = wm * 64 + mt * 16 + l16;
      bf16x8 lo = *(const bf16x8*)&Alds[r * LDSS + 8 * h];       // K: 8h..8h+7
      bf16x8 hi = *(const bf16x8*)&Alds[r * LDSS + 16 + 8 * h];  // K: 16+8h..
      afrag[mt] = __builtin_shufflevector(lo, hi, 0, 1, 2, 3, 4, 5, 6, 7,
                                          8, 9, 10, 11, 12, 13, 14, 15);
    }
    bf16x16 bfrag[2];
    #pragma unroll
    for (int nt = 0; nt < 2; ++nt) {
      int c = wn * 32 + nt * 16 + l16;
      bf16x8 lo = *(const bf16x8*)&Blds[c * LDSS + 8 * h];       // K: 8h..8h+7
      bf16x8 hi = *(const bf16x8*)&Blds[c * LDSS + 16 + 8 * h];  // K: 16+8h..
      bfrag[nt] = __builtin_shufflevector(lo, hi, 0, 1, 2, 3, 4, 5, 6, 7,
                                          8, 9, 10, 11, 12, 13, 14, 15);
    }
    #pragma unroll
    for (int mt = 0; mt < 4; ++mt)
      #pragma unroll
      for (int nt = 0; nt < 2; ++nt)
        acc[mt][nt] = __builtin_amdgcn_wmma_f32_16x16x32_bf16(
            false, afrag[mt], false, bfrag[nt], (short)0, acc[mt][nt],
            false, false);
  };

  const int nkt = K / BK;     // K is always a multiple of 32; nkt >= 8
  loadA(0);
  loadB(0);

  for (int kt = 0; kt < nkt - 1; ++kt) {
    storeTiles();             // staged regs -> LDS (regs dead afterwards)
    __syncthreads();
    int k0n = (kt + 1) * BK;  // unconditional prefetch; waits land next iter
    loadA(k0n);
    loadB(k0n);
    compute();
    __syncthreads();          // protect LDS before next store
  }
  storeTiles();               // peeled last tile: no prefetch
  __syncthreads();
  compute();

  // ---- epilogue: bias add + store (C/D layout: elem r -> M = r + 8h) ----
  #pragma unroll
  for (int nt = 0; nt < 2; ++nt) {
    int gn = n0 + wn * 32 + nt * 16 + l16;
    if (gn < N) {
      float bv = bias[gn];
      #pragma unroll
      for (int mt = 0; mt < 4; ++mt) {
        int gm = m0 + wm * 64 + mt * 16 + 8 * h;
        #pragma unroll
        for (int r = 0; r < 8; ++r)
          C[(size_t)(gm + r) * N + gn] = acc[mt][nt][r] + bv;
      }
    }
  }
}

// ---------------------------------------------------------------------------
// Host: 13 stream-ordered GEMM launches tracing the fusion DAG.
// ---------------------------------------------------------------------------

extern "C" void kernel_launch(void* const* d_in, const int* in_sizes, int n_in,
                              void* d_out, int out_size, void* d_ws, size_t ws_size,
                              hipStream_t stream) {
  (void)in_sizes; (void)n_in; (void)out_size; (void)ws_size;
  const int B = 8192;
  const float* x0 = (const float*)d_in[0];   // [B,1024]
  const float* x1 = (const float*)d_in[1];   // [B,512]
  const float* x2 = (const float*)d_in[2];   // [B,256]

  float* out   = (float*)d_out;
  float* t_out = out;                        // [B,512]
  float* eff   = out + (size_t)B * 512;      // [B,19]

  float* ws     = (float*)d_ws;
  float* hid1   = ws;                          // [B,1024]
  float* hid2   = hid1 + (size_t)B * 1024;     // [B,256]
  float* out01  = hid2 + (size_t)B * 256;      // [B,512]
  float* out02  = out01 + (size_t)B * 512;     // [B,512]
  float* out12  = out02 + (size_t)B * 512;     // [B,512]
  float* out012 = out12 + (size_t)B * 512;     // [B,512]  (~109 MB total)

  auto launch = [&](const SegList& sl, int widx, float* Cp, int N, int K) {
    const float* W = (const float*)d_in[widx];
    const float* b = (const float*)d_in[widx + 1];
    dim3 grid((unsigned)((N + BN - 1) / BN), (unsigned)(B / BM));
    hipLaunchKernelGGL(dfg_gemm_bf16_wmma, grid, dim3(256), 0, stream,
                       sl, W, b, Cp, (const float*)eff, N, K);
  };

  SegList s{};
  // --- efficacy MLP (no scaling) ---
  s.n = 3; s.s[0] = {x0, 1024, -1}; s.s[1] = {x1, 512, -1}; s.s[2] = {x2, 256, -1};
  launch(s, 3, hid1, 512, 1792);
  s.n = 1; s.s[0] = {hid1, 512, -1};
  launch(s, 5, hid2, 256, 512);
  s.n = 1; s.s[0] = {hid2, 256, -1};
  launch(s, 7, eff, 19, 256);

  // --- pair (0,1): eff cols 0,1 ---
  s.n = 2; s.s[0] = {x0, 1024, 0}; s.s[1] = {x1, 512, 1};
  launch(s, 9, hid1, 1024, 1536);
  s.n = 1; s.s[0] = {hid1, 1024, -1};
  launch(s, 11, out01, 512, 1024);

  // --- pair (0,2): eff cols 2,3 ---
  s.n = 2; s.s[0] = {x0, 1024, 2}; s.s[1] = {x2, 256, 3};
  launch(s, 13, hid1, 1024, 1280);
  s.n = 1; s.s[0] = {hid1, 1024, -1};
  launch(s, 15, out02, 512, 1024);

  // --- pair (1,2): eff cols 4,5 ---
  s.n = 2; s.s[0] = {x1, 512, 4}; s.s[1] = {x2, 256, 5};
  launch(s, 17, hid1, 1024, 768);
  s.n = 1; s.s[0] = {hid1, 1024, -1};
  launch(s, 19, out12, 512, 1024);

  // --- triple (0,1,2): eff cols 6..11 ---
  s.n = 6;
  s.s[0] = {x0, 1024, 6};  s.s[1] = {x1, 512, 7};   s.s[2] = {x2, 256, 8};
  s.s[3] = {out01, 512, 9}; s.s[4] = {out02, 512, 10}; s.s[5] = {out12, 512, 11};
  launch(s, 21, hid1, 1024, 3328);
  s.n = 1; s.s[0] = {hid1, 1024, -1};
  launch(s, 23, out012, 512, 1024);

  // --- t network: eff cols 12..18 ---
  s.n = 7;
  s.s[0] = {x0, 1024, 12};  s.s[1] = {x1, 512, 13};   s.s[2] = {x2, 256, 14};
  s.s[3] = {out01, 512, 15}; s.s[4] = {out02, 512, 16}; s.s[5] = {out12, 512, 17};
  s.s[6] = {out012, 512, 18};
  launch(s, 25, hid1, 1024, 3840);
  s.n = 1; s.s[0] = {hid1, 1024, -1};
  launch(s, 27, t_out, 512, 1024);
}
